// MaskedLinear_89721866814284
// MI455X (gfx1250) — matrix operand order repaired and checked by reading, not compiled, not run on hardware
//
#include <hip/hip_runtime.h>

#define IN_F   4096
#define OUT_F  4096
#define TOKENS 8192

typedef _Float16 half_t;
typedef __attribute__((ext_vector_type(16))) _Float16 v16h;
typedef __attribute__((ext_vector_type(8)))  _Float16 v8h;
typedef __attribute__((ext_vector_type(4)))  _Float16 v4h;
typedef __attribute__((ext_vector_type(8)))  float    v8f;

union F16Frag { v16h v; v8h h[2]; };

// ---------------------------------------------------------------- convert x -> f16
__global__ __launch_bounds__(256) void k_convert_x(const float* __restrict__ x,
                                                   half_t* __restrict__ xh, int n) {
    int i = (blockIdx.x * 256 + threadIdx.x) * 4;
    if (i < n) {
        float4 f = *(const float4*)(x + i);
        v4h h = { (half_t)f.x, (half_t)f.y, (half_t)f.z, (half_t)f.w };
        *(v4h*)(xh + i) = h;
    }
}

// ------------------------------------------------- ordered scatter (last write wins)
// key = ((i+1) << 32) | f32bits ; atomicMax picks the largest source index i.
__global__ __launch_bounds__(256) void k_scatter_ordered(const float* __restrict__ w,
                                                         const int* __restrict__ ri,
                                                         const int* __restrict__ ci,
                                                         unsigned long long* __restrict__ keys,
                                                         int nnz) {
    int i = blockIdx.x * 256 + threadIdx.x;
    if (i < nnz) {
        size_t slot = (size_t)ri[i] * IN_F + (size_t)ci[i];
        unsigned long long key =
            ((unsigned long long)(unsigned)(i + 1) << 32) | (unsigned long long)__float_as_uint(w[i]);
        atomicMax(keys + slot, key);
    }
}

__global__ __launch_bounds__(256) void k_decode(const unsigned long long* __restrict__ keys,
                                                half_t* __restrict__ wh, int n) {
    int i = blockIdx.x * 256 + threadIdx.x;
    if (i < n) {
        unsigned long long k = keys[i];
        wh[i] = k ? (half_t)__uint_as_float((unsigned)k) : (half_t)0.0f;
    }
}

// fallback if workspace too small for ordered keys (dup winner then unordered)
__global__ __launch_bounds__(256) void k_scatter_plain(const float* __restrict__ w,
                                                       const int* __restrict__ ri,
                                                       const int* __restrict__ ci,
                                                       half_t* __restrict__ wh, int nnz) {
    int i = blockIdx.x * 256 + threadIdx.x;
    if (i < nnz) {
        size_t slot = (size_t)ri[i] * IN_F + (size_t)ci[i];
        wh[slot] = (half_t)w[i];
    }
}

// ---------------------------------------------------------------- WMMA GEMM
// out[M=8192, N=4096] = Xh[M,K] * Wh[N,K]^T + bias[N]
// One wave computes a 64x64 block as a 4x4 grid of 16x16 WMMA tiles.
// Block = 128 threads = 4 waves stacked along M; grid = (N/64, M/256).
__global__ __launch_bounds__(128) void k_gemm(const half_t* __restrict__ Xh,
                                              const half_t* __restrict__ Wh,
                                              const float* __restrict__ bias,
                                              float* __restrict__ out) {
    const int lane = threadIdx.x & 31;
    const int wave = threadIdx.x >> 5;
    const int row  = lane & 15;   // M row (A) / N column (B) within tile
    const int hi   = lane >> 4;   // half-wave selector for K striping

    const int n_base = blockIdx.x * 64;
    const int m_base = (blockIdx.y * 4 + wave) * 64;

    // Per-lane base pointers implementing the ISA VGPR layouts:
    // A (16x32 f16): lane<16 -> K {0..7, 16..23}; lane>=16 -> K {8..15, 24..31}
    // B (32x16 f16): lane holds column (lane&15); lane<16 -> K {0..7, 16..23}; lane>=16 -> K {8..15, 24..31}
    const half_t* aptr[4];
    const half_t* bptr[4];
#pragma unroll
    for (int mt = 0; mt < 4; ++mt)
        aptr[mt] = Xh + (size_t)(m_base + mt * 16 + row) * IN_F + hi * 8;
#pragma unroll
    for (int nt = 0; nt < 4; ++nt)
        bptr[nt] = Wh + (size_t)(n_base + nt * 16 + row) * IN_F + hi * 8;

    // Accumulators initialized with bias (all 8 M-rows of a tile share one N per lane)
    v8f acc[4][4];
#pragma unroll
    for (int nt = 0; nt < 4; ++nt) {
        const float bv = bias[n_base + nt * 16 + row];
#pragma unroll
        for (int mt = 0; mt < 4; ++mt) {
#pragma unroll
            for (int e = 0; e < 8; ++e) acc[mt][nt][e] = bv;
        }
    }

    for (int k0 = 0; k0 < IN_F; k0 += 32) {
        F16Frag a[4];
#pragma unroll
        for (int mt = 0; mt < 4; ++mt) {
            a[mt].h[0] = *(const v8h*)(aptr[mt] + k0);
            a[mt].h[1] = *(const v8h*)(aptr[mt] + k0 + 16);
        }
#pragma unroll
        for (int nt = 0; nt < 4; ++nt) {
            F16Frag b;
            b.h[0] = *(const v8h*)(bptr[nt] + k0);
            b.h[1] = *(const v8h*)(bptr[nt] + k0 + 16);
#pragma unroll
            for (int mt = 0; mt < 4; ++mt)
                acc[mt][nt] = __builtin_amdgcn_wmma_f32_16x16x32_f16(
                    false, a[mt].v, false, b.v, (short)0, acc[mt][nt], false, false);
        }
    }

    // Store: C/D layout -> M = tile_m + r + 8*hi, N = tile_n + (lane&15)
#pragma unroll
    for (int mt = 0; mt < 4; ++mt) {
#pragma unroll
        for (int nt = 0; nt < 4; ++nt) {
            const int n = n_base + nt * 16 + row;
            float* o = out + (size_t)(m_base + mt * 16 + 8 * hi) * OUT_F + n;
#pragma unroll
            for (int r = 0; r < 8; ++r)
                o[(size_t)r * OUT_F] = acc[mt][nt][r];
        }
    }
}

extern "C" void kernel_launch(void* const* d_in, const int* in_sizes, int n_in,
                              void* d_out, int out_size, void* d_ws, size_t ws_size,
                              hipStream_t stream) {
    const float* x    = (const float*)d_in[0];
    const float* w    = (const float*)d_in[1];
    const float* bias = (const float*)d_in[2];
    const int*   ri   = (const int*)d_in[3];
    const int*   ci   = (const int*)d_in[4];
    const int    nnz  = in_sizes[1];
    float*       out  = (float*)d_out;

    char* base = (char*)d_ws;
    const size_t xh_bytes  = (size_t)TOKENS * IN_F * sizeof(half_t);   //  64 MB
    const size_t wh_bytes  = (size_t)OUT_F  * IN_F * sizeof(half_t);   //  32 MB
    const size_t key_bytes = (size_t)OUT_F  * IN_F * 8;                // 128 MB
    half_t* Xh = (half_t*)base;
    half_t* Wh = (half_t*)(base + xh_bytes);
    unsigned long long* keys = (unsigned long long*)(base + xh_bytes + wh_bytes);

    // 1) x -> f16
    {
        const int n = TOKENS * IN_F;
        k_convert_x<<<n / (256 * 4), 256, 0, stream>>>(x, Xh, n);
    }

    // 2) scatter weights into dense f16 W
    if (ws_size >= xh_bytes + wh_bytes + key_bytes) {
        hipMemsetAsync(keys, 0, key_bytes, stream);
        k_scatter_ordered<<<(nnz + 255) / 256, 256, 0, stream>>>(w, ri, ci, keys, nnz);
        k_decode<<<(OUT_F * IN_F) / 256, 256, 0, stream>>>(keys, Wh, OUT_F * IN_F);
    } else {
        hipMemsetAsync(Wh, 0, wh_bytes, stream);
        k_scatter_plain<<<(nnz + 255) / 256, 256, 0, stream>>>(w, ri, ci, Wh, nnz);
    }

    // 3) WMMA GEMM + bias
    dim3 grid(OUT_F / 64, TOKENS / 256);
    k_gemm<<<grid, 128, 0, stream>>>(Xh, Wh, bias, out);

    (void)n_in; (void)out_size; (void)ws_size;
}
